// RGLRU_11338713662140
// MI455X (gfx1250) — matrix operand order repaired and checked by reading, not compiled, not run on hardware
//
#include <hip/hip_runtime.h>

// RG-LRU fused: WMMA f16 gate GEMMs + elementwise epilogue + in-place scan.
// MI455X / gfx1250, wave32. Workspace use: ~35 MB.

typedef __attribute__((ext_vector_type(16))) _Float16 v16h;
typedef __attribute__((ext_vector_type(8)))  float    v8f;

#define RG_H  16
#define RG_W  2048
#define RG_BW 128
#define RG_B  4
#define RG_S  2048
#define RG_BS (RG_B * RG_S)

// ws layout:
//   [0, 1MB)            : f16 B-fragments for both gate weight tensors
//   [1MB, 1MB+8KB)      : softplus(recurrent_param), fp32 [W]
//   [1MB+64KB, +32MB)   : per-element decay 'a' as f16 [B*S*W]
#define BFRAG_HALFS (2 * 16 * 8 * 4 * 32 * 16)   // 524288
#define SP_OFF_BYTES ((size_t)BFRAG_HALFS * 2)   // 1 MiB
#define AH_OFF_BYTES (SP_OFF_BYTES + 65536)

// ---------------------------------------------------------------------------
// Kernel 0: pack gate weights into WMMA B-fragment order (f16) + softplus(rp).
// B-matrix 32x16 f16 layout used here (contiguous k, same as A): lanes 0-15 hold N=lane, K=0..15
// (half j == K); lanes 16-31 hold N=lane-16, K=16..31 (half j == K-16).
// Fragment storage: (((g*16+h)*8+jt)*4+c)*512 + lane*16 + j  (halfs)
// ---------------------------------------------------------------------------
__global__ __launch_bounds__(256) void rg_prep_kernel(
    const float* __restrict__ igw, const float* __restrict__ rgw,
    const float* __restrict__ rp, _Float16* __restrict__ bfrag,
    float* __restrict__ sp) {
  int gid = blockIdx.x * blockDim.x + threadIdx.x;
  if (gid < BFRAG_HALFS) {
    int j    = gid & 15;
    int lane = (gid >> 4) & 31;
    int c    = (gid >> 9) & 3;
    int jt   = (gid >> 11) & 7;
    int h    = (gid >> 14) & 15;
    int g    = (gid >> 18) & 1;
    int k = c * 32 + ((lane >> 4) ? 16 : 0) + j;   // K index 0..127
    int n = jt * 16 + (lane & 15);                 // output channel 0..127
    const float* wsrc = g ? rgw : igw;
    bfrag[gid] = (_Float16)wsrc[((size_t)h * RG_BW + k) * RG_BW + n];
  }
  if (gid < RG_W) {
    float r = rp[gid];
    sp[gid] = (r > 20.f) ? r : log1pf(expf(r));    // softplus
  }
}

// ---------------------------------------------------------------------------
// Kernel 1: per (16-token tile, head) workgroup, 256 threads = 8 waves.
// Wave jt computes one 16x16 column tile of both gates via 4x
// v_wmma_f32_16x16x32_f16 each, then fused elementwise epilogue.
// Writes normalized_x (fp32) into d_out and decay a (f16) into ws.
// ---------------------------------------------------------------------------
__global__ __launch_bounds__(256) void rg_gates_kernel(
    const float* __restrict__ act, const int* __restrict__ pos,
    const float* __restrict__ igb, const float* __restrict__ rgb,
    const _Float16* __restrict__ bfrag, const float* __restrict__ sp,
    float* __restrict__ xout, _Float16* __restrict__ aout) {
  __shared__ float sAct[16][RG_BW + 4];                      // padded vs bank conflicts
  __shared__ __align__(32) _Float16 sAfrag[4][32][16];       // A-fragments, 4 KB

  const int tid = threadIdx.x;
  const int n0  = blockIdx.x * 16;   // first global token of tile
  const int h   = blockIdx.y;        // head

  // Cooperative load: 16 tokens x 128 floats of head h; scatter into the
  // A-fragment layout used here (16-bit A 16x32, contiguous k: lane half h holds k=16h+j).
  {
    const int r  = tid >> 4;            // token row in tile
    const int c0 = (tid & 15) * 8;      // channel base
    const float* g = act + (size_t)(n0 + r) * RG_W + h * RG_BW + c0;
    float4 v0 = *(const float4*)(g);
    float4 v1 = *(const float4*)(g + 4);
    float vals[8] = {v0.x, v0.y, v0.z, v0.w, v1.x, v1.y, v1.z, v1.w};
#pragma unroll
    for (int i = 0; i < 8; ++i) {
      int k = c0 + i;
      sAct[r][k] = vals[i];
      int cc = k >> 5;
      int rr = k & 31;
      int lane = r + (((rr >> 4) & 1) ? 16 : 0);
      int jj   = rr & 15;
      sAfrag[cc][lane][jj] = (_Float16)vals[i];
    }
  }
  __syncthreads();

  const int wave = tid >> 5;   // = jt, output column tile
  const int lane = tid & 31;

  v8f acc_i = {};
  v8f acc_r = {};
#pragma unroll
  for (int c = 0; c < 4; ++c) {
    v16h a  = *(const v16h*)(&sAfrag[c][lane][0]);
    v16h bi = *(const v16h*)(bfrag + ((size_t)((((0 * 16 + h) * 8 + wave) * 4 + c) * 32 + lane)) * 16);
    v16h br = *(const v16h*)(bfrag + ((size_t)((((1 * 16 + h) * 8 + wave) * 4 + c) * 32 + lane)) * 16);
    acc_i = __builtin_amdgcn_wmma_f32_16x16x32_f16(false, a, false, bi,
                                                   (short)0, acc_i, false, false);
    acc_r = __builtin_amdgcn_wmma_f32_16x16x32_f16(false, a, false, br,
                                                   (short)0, acc_r, false, false);
  }

  // C/D layout: VGPR i -> M=i (lanes 0-15), M=8+i (lanes 16-31); N = lane%16.
  const int ncol  = wave * 16 + (lane & 15);
  const int widx  = h * RG_BW + ncol;
  const float bI  = igb[widx];
  const float bR  = rgb[widx];
  const float spv = sp[widx];
  const int mbase = (lane >> 4) ? 8 : 0;
#pragma unroll
  for (int i = 0; i < 8; ++i) {
    int m   = i + mbase;
    int tok = n0 + m;
    float ig = 1.f / (1.f + expf(-(acc_i[i] + bI)));
    float rg = 1.f / (1.f + expf(-(acc_r[i] + bR)));
    float log_a = -8.f * rg * spv;
    float a  = expf(log_a);
    float a2 = a * a;                         // exp(2*log_a)
    float mult = sqrtf(fmaxf(1.f - a2, 0.f)); // sqrt_bound fwd
    if (pos[tok] == 0) { mult = 1.f; a = 0.f; }   // reset at doc boundary
    float x = sAct[m][ncol] * ig * mult;
    size_t o = (size_t)tok * RG_W + widx;
    xout[o] = x;
    aout[o] = (_Float16)a;
  }
}

// ---------------------------------------------------------------------------
// Kernel 2: linear recurrence h_t = a_t*h_{t-1} + x_t, in place on d_out.
// One thread per (b,w) channel; coalesced across the wave each timestep.
// ---------------------------------------------------------------------------
__global__ __launch_bounds__(256) void rg_scan_kernel(
    const _Float16* __restrict__ ah, float* __restrict__ out) {
  int ch = blockIdx.x * blockDim.x + threadIdx.x;  // 0..B*W-1
  int b = ch >> 11;          // / W (W == 2048)
  int w = ch & (RG_W - 1);
  size_t base = (size_t)b * RG_S * RG_W + w;
  float hacc = 0.f;
  for (int t = 0; t < RG_S; t += 4) {
    size_t i0 = base + (size_t)t * RG_W;
    if (t + 8 < RG_S) {                         // speculative prefetch ahead
      __builtin_prefetch(out + i0 + (size_t)8 * RG_W, 0, 1);
      __builtin_prefetch(ah  + i0 + (size_t)8 * RG_W, 0, 1);
    }
    float a0 = (float)ah[i0];
    float a1 = (float)ah[i0 + RG_W];
    float a2 = (float)ah[i0 + 2 * RG_W];
    float a3 = (float)ah[i0 + 3 * RG_W];
    float x0 = out[i0];
    float x1 = out[i0 + RG_W];
    float x2 = out[i0 + 2 * RG_W];
    float x3 = out[i0 + 3 * RG_W];
    hacc = fmaf(a0, hacc, x0); out[i0] = hacc;
    hacc = fmaf(a1, hacc, x1); out[i0 + RG_W] = hacc;
    hacc = fmaf(a2, hacc, x2); out[i0 + 2 * RG_W] = hacc;
    hacc = fmaf(a3, hacc, x3); out[i0 + 3 * RG_W] = hacc;
  }
}

extern "C" void kernel_launch(void* const* d_in, const int* in_sizes, int n_in,
                              void* d_out, int out_size, void* d_ws, size_t ws_size,
                              hipStream_t stream) {
  const float* act = (const float*)d_in[0];
  const int*   pos = (const int*)d_in[1];
  const float* rp  = (const float*)d_in[2];
  const float* igw = (const float*)d_in[3];
  const float* igb = (const float*)d_in[4];
  const float* rgw = (const float*)d_in[5];
  const float* rgb = (const float*)d_in[6];
  float* out = (float*)d_out;

  char* ws = (char*)d_ws;
  _Float16* bfrag = (_Float16*)ws;
  float*    sp    = (float*)(ws + SP_OFF_BYTES);
  _Float16* ah    = (_Float16*)(ws + AH_OFF_BYTES);

  rg_prep_kernel<<<BFRAG_HALFS / 256, 256, 0, stream>>>(igw, rgw, rp, bfrag, sp);

  dim3 g1(RG_BS / 16, RG_H);
  rg_gates_kernel<<<g1, 256, 0, stream>>>(act, pos, igb, rgb, bfrag, sp, out, ah);

  rg_scan_kernel<<<(RG_B * RG_W) / 256, 256, 0, stream>>>(ah, out);
}
